// MultiheadAttention_4123168604638
// MI455X (gfx1250) — matrix operand order repaired and checked by reading, not compiled, not run on hardware
//
#include <hip/hip_runtime.h>
#include <hip/hip_bf16.h>

#define BATCH 4
#define SEQ   2048
#define DIM   256
#define HEADS 8
#define HDIM  32
#define LSEL  1024
#define QK_SCALE 0.17677669529663687f  /* 1/sqrt(32) */

typedef __attribute__((ext_vector_type(16))) __bf16 v16bf;
typedef __attribute__((ext_vector_type(8)))  float  v8f;

union FragB {
  v16bf v;
  __bf16 h[16];
  unsigned short u[16];
  uint4 q[2];
};

__device__ __forceinline__ unsigned short f2bf(float f) {
  return __builtin_bit_cast(unsigned short, (__bf16)f);   // native v_cvt (RNE)
}

__device__ __forceinline__ v8f wmma_bf16(const v16bf& a, const v16bf& b, const v8f& c) {
  return __builtin_amdgcn_wmma_f32_16x16x32_bf16(false, a, false, b, (short)0, c, false, false);
}

// A-matrix (16x32 bf16): per lane 16 contiguous K elements.
__device__ __forceinline__ void load_fragA(FragB& f, const unsigned short* p) {
  f.q[0] = *(const uint4*)(p);
  f.q[1] = *(const uint4*)(p + 8);
}
// B-matrix (32x16 bf16): per lane 16 contiguous K elements.
__device__ __forceinline__ void load_fragB(FragB& f, const unsigned short* p) {
  f.q[0] = *(const uint4*)(p);
  f.q[1] = *(const uint4*)(p + 8);
}
// Build a fragment half (8 bf16) from 8 f32 loaded as two float4.
__device__ __forceinline__ void cvt8(FragB& f, int base, const float* p) {
  const float4 a = *(const float4*)(p);
  const float4 b = *(const float4*)(p + 4);
  f.h[base + 0] = (__bf16)a.x; f.h[base + 1] = (__bf16)a.y;
  f.h[base + 2] = (__bf16)a.z; f.h[base + 3] = (__bf16)a.w;
  f.h[base + 4] = (__bf16)b.x; f.h[base + 5] = (__bf16)b.y;
  f.h[base + 6] = (__bf16)b.z; f.h[base + 7] = (__bf16)b.w;
}

/* ---------------- 0: zero weight accumulator ---------------- */
__global__ void k_zero(float* __restrict__ p, int n) {
  int i = blockIdx.x * blockDim.x + threadIdx.x;
  if (i < n) p[i] = 0.0f;
}

/* ---------------- 1: qkv projection -> qh/kh/vT bf16 ---------------- */
__global__ __launch_bounds__(32) void k_qkv(const float* __restrict__ Q,
    const float* __restrict__ Win, const float* __restrict__ bin,
    unsigned short* __restrict__ qh, unsigned short* __restrict__ kh,
    unsigned short* __restrict__ vT)
{
  const int lane = threadIdx.x, l16 = lane & 15, hi = lane >> 4;
  const int bid = blockIdx.x;
  const int mt = bid % ((BATCH * SEQ) / 16);
  const int nt = bid / ((BATCH * SEQ) / 16);
  const int m0 = mt * 16, n0 = nt * 32;

  v8f acc0 = {}, acc1 = {};
#pragma unroll
  for (int k0 = 0; k0 < DIM; k0 += 32) {
    FragB a;
    const float* ap = Q + (size_t)(m0 + l16) * DIM + k0 + (hi ? 16 : 0);
    cvt8(a, 0, ap);
    cvt8(a, 8, ap + 8);
    FragB b0, b1;
    const float* bp0 = Win + (size_t)(n0 + l16) * DIM + k0 + (hi ? 16 : 0);
    const float* bp1 = Win + (size_t)(n0 + 16 + l16) * DIM + k0 + (hi ? 16 : 0);
    cvt8(b0, 0, bp0); cvt8(b0, 8, bp0 + 8);
    cvt8(b1, 0, bp1); cvt8(b1, 8, bp1 + 8);
    acc0 = wmma_bf16(a.v, b0.v, acc0);
    acc1 = wmma_bf16(a.v, b1.v, acc1);
  }
#pragma unroll
  for (int r = 0; r < 8; ++r) {
    const int row = m0 + r + hi * 8;
    const int b = row / SEQ, t = row % SEQ;
#pragma unroll
    for (int sub = 0; sub < 2; ++sub) {
      const int f = n0 + sub * 16 + l16;
      const float val = (sub ? acc1[r] : acc0[r]) + bin[f];
      if (f < DIM) {
        const int h = f >> 5, c = f & 31;
        qh[(((size_t)b * HEADS + h) * SEQ + t) * HDIM + c] = f2bf(val * QK_SCALE);
      } else if (f < 2 * DIM) {
        const int ff = f - DIM, h = ff >> 5, c = ff & 31;
        kh[(((size_t)b * HEADS + h) * SEQ + t) * HDIM + c] = f2bf(val);
      } else {
        const int ff = f - 2 * DIM, h = ff >> 5, c = ff & 31;
        vT[(((size_t)b * HEADS + h) * HDIM + c) * SEQ + t] = f2bf(val);
      }
    }
  }
}

/* ---------------- 2: softmax row stats (max, sumexp) ---------------- */
__global__ __launch_bounds__(32) void k_stats(const unsigned short* __restrict__ qh,
    const unsigned short* __restrict__ kh,
    float* __restrict__ mArr, float* __restrict__ sArr)
{
  const int lane = threadIdx.x, l16 = lane & 15, hi = lane >> 4;
  const int bid = blockIdx.x;
  const int qt = bid % (SEQ / 16);
  const int h  = (bid / (SEQ / 16)) % HEADS;
  const int b  = bid / ((SEQ / 16) * HEADS);
  const size_t bh = (size_t)b * HEADS + h;

  FragB a;
  load_fragA(a, qh + (bh * SEQ + qt * 16 + l16) * HDIM + (hi ? 16 : 0));

  const v8f zero = {};
  float mmax[8];
#pragma unroll
  for (int r = 0; r < 8; ++r) mmax[r] = -3.0e38f;
  for (int jt = 0; jt < SEQ / 16; ++jt) {
    FragB kb;
    load_fragB(kb, kh + (bh * SEQ + jt * 16 + l16) * HDIM + (hi ? 16 : 0));
    v8f s = wmma_bf16(a.v, kb.v, zero);
#pragma unroll
    for (int r = 0; r < 8; ++r) mmax[r] = fmaxf(mmax[r], s[r]);
  }
#pragma unroll
  for (int r = 0; r < 8; ++r)
    for (int msk = 1; msk < 16; msk <<= 1)
      mmax[r] = fmaxf(mmax[r], __shfl_xor(mmax[r], msk, 32));

  float ssum[8];
#pragma unroll
  for (int r = 0; r < 8; ++r) ssum[r] = 0.0f;
  for (int jt = 0; jt < SEQ / 16; ++jt) {
    FragB kb;
    load_fragB(kb, kh + (bh * SEQ + jt * 16 + l16) * HDIM + (hi ? 16 : 0));
    v8f s = wmma_bf16(a.v, kb.v, zero);
#pragma unroll
    for (int r = 0; r < 8; ++r) ssum[r] += __expf(s[r] - mmax[r]);
  }
#pragma unroll
  for (int r = 0; r < 8; ++r)
    for (int msk = 1; msk < 16; msk <<= 1)
      ssum[r] += __shfl_xor(ssum[r], msk, 32);

  if (l16 == 0) {
#pragma unroll
    for (int r = 0; r < 8; ++r) {
      const int row = qt * 16 + r + hi * 8;
      mArr[bh * SEQ + row] = mmax[r];
      sArr[bh * SEQ + row] = ssum[r];
    }
  }
}

/* ---------------- 3: weight[b,j] = sum_i max_h A ---------------- */
__global__ __launch_bounds__(32) void k_weight(const unsigned short* __restrict__ qh,
    const unsigned short* __restrict__ kh,
    const float* __restrict__ mArr, const float* __restrict__ sArr,
    float* __restrict__ weight)
{
  const int lane = threadIdx.x, l16 = lane & 15, hi = lane >> 4;
  const int bid = blockIdx.x;
  const int qg = bid & 7;
  const int jt = (bid >> 3) & 127;
  const int b  = bid >> 10;
  const v8f zero = {};

  float wsum = 0.0f;
  for (int qi = 0; qi < 16; ++qi) {
    const int qt = qg * 16 + qi;
    float amax[8];
#pragma unroll
    for (int r = 0; r < 8; ++r) amax[r] = 0.0f;
    for (int h = 0; h < HEADS; ++h) {
      const size_t bh = (size_t)b * HEADS + h;
      FragB a, kb;
      load_fragA(a, qh + (bh * SEQ + qt * 16 + l16) * HDIM + (hi ? 16 : 0));
      load_fragB(kb, kh + (bh * SEQ + jt * 16 + l16) * HDIM + (hi ? 16 : 0));
      v8f s = wmma_bf16(a.v, kb.v, zero);
#pragma unroll
      for (int r = 0; r < 8; ++r) {
        const int row = qt * 16 + r + hi * 8;
        const float m  = mArr[bh * SEQ + row];
        const float sv = sArr[bh * SEQ + row];
        const float p = __expf(s[r] - m) / sv;
        amax[r] = fmaxf(amax[r], p);
      }
    }
#pragma unroll
    for (int r = 0; r < 8; ++r) wsum += amax[r];
  }
  wsum += __shfl_xor(wsum, 16, 32);
  if (!hi) atomicAdd(&weight[(size_t)b * SEQ + jt * 16 + l16], wsum);
}

/* ---------------- 4: per-batch top-k (bitonic sort, desc val / asc idx) ---------------- */
__global__ __launch_bounds__(1024) void k_topk(const float* __restrict__ weight,
    int* __restrict__ inds, float* __restrict__ outW, int* __restrict__ outI)
{
  __shared__ float vals[SEQ];
  __shared__ int   idxs[SEQ];
  const int b = blockIdx.x, tid = threadIdx.x;
  for (int i = tid; i < SEQ; i += 1024) { vals[i] = weight[(size_t)b * SEQ + i]; idxs[i] = i; }
  __syncthreads();
  for (int k = 2; k <= SEQ; k <<= 1) {
    for (int j = k >> 1; j > 0; j >>= 1) {
      for (int i = tid; i < SEQ; i += 1024) {
        const int ixj = i ^ j;
        if (ixj > i) {
          const float v1 = vals[i], v2 = vals[ixj];
          const int i1 = idxs[i], i2 = idxs[ixj];
          const bool later = (v1 < v2) || (v1 == v2 && i1 > i2);
          const bool desc = ((i & k) == 0);
          if (desc ? later : !later) {
            vals[i] = v2; vals[ixj] = v1; idxs[i] = i2; idxs[ixj] = i1;
          }
        }
      }
      __syncthreads();
    }
  }
  for (int i = tid; i < LSEL; i += 1024) { inds[b * LSEL + i] = idxs[i]; outI[b * LSEL + i] = idxs[i]; }
  for (int i = tid; i < SEQ; i += 1024) outW[(size_t)b * SEQ + i] = weight[(size_t)b * SEQ + i];
}

/* ---------------- 5: O = A[selected] @ v (recomputed, flash-style) ---------------- */
__global__ __launch_bounds__(32) void k_attnout(const unsigned short* __restrict__ qh,
    const unsigned short* __restrict__ kh, const unsigned short* __restrict__ vT,
    const float* __restrict__ mArr, const float* __restrict__ sArr,
    const int* __restrict__ inds, unsigned short* __restrict__ Otmp)
{
  __shared__ unsigned short P[16 * 32];
  const int lane = threadIdx.x, l16 = lane & 15, hi = lane >> 4;
  const int bid = blockIdx.x;
  const int lt = bid % (LSEL / 16);
  const int h  = (bid / (LSEL / 16)) % HEADS;
  const int b  = bid / ((LSEL / 16) * HEADS);
  const size_t bh = (size_t)b * HEADS + h;

  float mrow[8], sinv[8];
#pragma unroll
  for (int r = 0; r < 8; ++r) {
    const int l = lt * 16 + r + hi * 8;
    const int q = inds[b * LSEL + l];
    mrow[r] = mArr[bh * SEQ + q];
    sinv[r] = 1.0f / sArr[bh * SEQ + q];
  }
  const int qa = inds[b * LSEL + lt * 16 + l16];
  FragB a;
  load_fragA(a, qh + (bh * SEQ + qa) * HDIM + (hi ? 16 : 0));

  const v8f zero = {};
  v8f o0 = {}, o1 = {};
  for (int j0 = 0; j0 < SEQ; j0 += 32) {
    FragB k0f, k1f;
    load_fragB(k0f, kh + (bh * SEQ + j0 + l16) * HDIM + (hi ? 16 : 0));
    load_fragB(k1f, kh + (bh * SEQ + j0 + 16 + l16) * HDIM + (hi ? 16 : 0));
    v8f s0 = wmma_bf16(a.v, k0f.v, zero);
    v8f s1 = wmma_bf16(a.v, k1f.v, zero);
#pragma unroll
    for (int r = 0; r < 8; ++r) {
      const int row = r + hi * 8;                       // C-layout: VGPR->row, lane->col
      P[row * 32 + l16]      = f2bf(__expf(s0[r] - mrow[r]) * sinv[r]);
      P[row * 32 + 16 + l16] = f2bf(__expf(s1[r] - mrow[r]) * sinv[r]);
    }
    asm volatile("s_wait_dscnt 0" ::: "memory");        // cross-lane LDS exchange
    FragB pa;
    load_fragA(pa, &P[l16 * 32 + (hi ? 16 : 0)]);
    FragB v0f, v1f;
    load_fragB(v0f, vT + (bh * HDIM + l16)      * SEQ + j0 + (hi ? 16 : 0));
    load_fragB(v1f, vT + (bh * HDIM + 16 + l16) * SEQ + j0 + (hi ? 16 : 0));
    o0 = wmma_bf16(pa.v, v0f.v, o0);
    o1 = wmma_bf16(pa.v, v1f.v, o1);
    asm volatile("s_wait_dscnt 0" ::: "memory");        // P fully consumed before overwrite
  }
#pragma unroll
  for (int r = 0; r < 8; ++r) {
    const int l = lt * 16 + r + hi * 8;
    unsigned short* op = Otmp + ((size_t)b * LSEL + l) * DIM + h * HDIM;
    op[l16]      = f2bf(o0[r]);
    op[16 + l16] = f2bf(o1[r]);
  }
}

/* ---------------- 6: out projection -> d_out ---------------- */
__global__ __launch_bounds__(32) void k_outproj(const unsigned short* __restrict__ Otmp,
    const float* __restrict__ Wout, const float* __restrict__ bout,
    float* __restrict__ outO)
{
  const int lane = threadIdx.x, l16 = lane & 15, hi = lane >> 4;
  const int bid = blockIdx.x;
  const int mt = bid % ((BATCH * LSEL) / 16);
  const int nt = bid / ((BATCH * LSEL) / 16);
  const int m0 = mt * 16, n0 = nt * 32;

  v8f acc0 = {}, acc1 = {};
#pragma unroll
  for (int k0 = 0; k0 < DIM; k0 += 32) {
    FragB a;
    load_fragA(a, Otmp + (size_t)(m0 + l16) * DIM + k0 + (hi ? 16 : 0));
    FragB b0, b1;
    const float* bp0 = Wout + (size_t)(n0 + l16) * DIM + k0 + (hi ? 16 : 0);
    const float* bp1 = Wout + (size_t)(n0 + 16 + l16) * DIM + k0 + (hi ? 16 : 0);
    cvt8(b0, 0, bp0); cvt8(b0, 8, bp0 + 8);
    cvt8(b1, 0, bp1); cvt8(b1, 8, bp1 + 8);
    acc0 = wmma_bf16(a.v, b0.v, acc0);
    acc1 = wmma_bf16(a.v, b1.v, acc1);
  }
#pragma unroll
  for (int r = 0; r < 8; ++r) {
    const int m = m0 + r + hi * 8;
    const int nA = n0 + l16, nB = n0 + 16 + l16;
    outO[(size_t)m * DIM + nA] = acc0[r] + bout[nA];
    outO[(size_t)m * DIM + nB] = acc1[r] + bout[nB];
  }
}

extern "C" void kernel_launch(void* const* d_in, const int* in_sizes, int n_in,
                              void* d_out, int out_size, void* d_ws, size_t ws_size,
                              hipStream_t stream)
{
  const float* Q    = (const float*)d_in[0];
  /* d_in[1] (K) and d_in[2] (V) are unused by the reference module (bug preserved) */
  const float* Win  = (const float*)d_in[3];
  const float* bin  = (const float*)d_in[4];
  const float* Wout = (const float*)d_in[5];
  const float* bout = (const float*)d_in[6];

  char* w = (char*)d_ws;
  unsigned short* qh = (unsigned short*)w; w += (size_t)BATCH * HEADS * SEQ * HDIM * 2;
  unsigned short* kh = (unsigned short*)w; w += (size_t)BATCH * HEADS * SEQ * HDIM * 2;
  unsigned short* vT = (unsigned short*)w; w += (size_t)BATCH * HEADS * SEQ * HDIM * 2;
  float* mArr   = (float*)w; w += (size_t)BATCH * HEADS * SEQ * 4;
  float* sArr   = (float*)w; w += (size_t)BATCH * HEADS * SEQ * 4;
  float* weight = (float*)w; w += (size_t)BATCH * SEQ * 4;
  int*   inds   = (int*)w;   w += (size_t)BATCH * LSEL * 4;
  unsigned short* Otmp = (unsigned short*)w; w += (size_t)BATCH * LSEL * DIM * 2;

  float* outO = (float*)d_out;                         // [B,L,D]
  float* outW = outO + (size_t)BATCH * LSEL * DIM;     // [B,T]
  int*   outI = (int*)(outW + (size_t)BATCH * SEQ);    // [B,L] int32 bits

  k_zero<<<(BATCH * SEQ + 255) / 256, 256, 0, stream>>>(weight, BATCH * SEQ);
  k_qkv<<<((BATCH * SEQ) / 16) * ((3 * DIM) / 32), 32, 0, stream>>>(Q, Win, bin, qh, kh, vT);
  k_stats<<<BATCH * HEADS * (SEQ / 16), 32, 0, stream>>>(qh, kh, mArr, sArr);
  k_weight<<<BATCH * 128 * 8, 32, 0, stream>>>(qh, kh, mArr, sArr, weight);
  k_topk<<<BATCH, 1024, 0, stream>>>(weight, inds, outW, outI);
  k_attnout<<<BATCH * HEADS * (LSEL / 16), 32, 0, stream>>>(qh, kh, vT, mArr, sArr, inds, Otmp);
  k_outproj<<<((BATCH * LSEL) / 16) * (DIM / 32), 32, 0, stream>>>(Otmp, Wout, bout, outO);
}
